// SwinTransformerBlock_89146341195940
// MI455X (gfx1250) — matrix operand order repaired and checked by reading, not compiled, not run on hardware
//
#include <hip/hip_runtime.h>

// ---------------- types ----------------
typedef _Float16 half_t;
typedef __attribute__((ext_vector_type(8)))  _Float16 v8h;
typedef __attribute__((ext_vector_type(16))) _Float16 v16h;
typedef __attribute__((ext_vector_type(8)))  float    v8f;

#define TOK 401408      // B * H * W = 128*56*56
#define NWIN 8192       // B * 64 windows
#define SCALE 0.17677669529663689f

__device__ __forceinline__ v16h combine8(v8h lo, v8h hi8) {
  v16h r;
#pragma unroll
  for (int i = 0; i < 8; ++i) { r[i] = lo[i]; r[i + 8] = hi8[i]; }
  return r;
}

__device__ __forceinline__ v8f wmma16(v16h a, v16h b, v8f c) {
  // D = A(16x32 f16) * B(32x16 f16) + C(16x16 f32)
  return __builtin_amdgcn_wmma_f32_16x16x32_f16(false, a, false, b, (short)0, c,
                                                false, false);
}

__device__ __forceinline__ int regionOf(int h) {  // shift-mask region id along one axis
  return h < 49 ? 0 : (h < 53 ? 1 : 2);
}

// ---------------- f32 -> f16 convert (weights) ----------------
__global__ void cvt_f16_kernel(const float* __restrict__ in, half_t* __restrict__ out, int n) {
  int i = blockIdx.x * 256 + threadIdx.x;
  if (i < n) out[i] = (half_t)in[i];
}

// ---------------- LayerNorm (+ optional roll & window partition) ----------------
// 1 wave = 1 token (96 channels, 3 per lane). doShift=1: write to window-major row.
__global__ __launch_bounds__(128) void ln_kernel(const float* __restrict__ x,
                                                 const float* __restrict__ g,
                                                 const float* __restrict__ b,
                                                 half_t* __restrict__ out, int doShift) {
  int lane = threadIdx.x & 31, wid = threadIdx.x >> 5;
  int t = blockIdx.x * 4 + wid;
  const float* xp = x + (size_t)t * 96;
  float v0 = xp[lane], v1 = xp[lane + 32], v2 = xp[lane + 64];
  float s = v0 + v1 + v2;
#pragma unroll
  for (int m = 16; m; m >>= 1) s += __shfl_xor(s, m, 32);
  float mean = s * (1.0f / 96.0f);
  float d0 = v0 - mean, d1 = v1 - mean, d2 = v2 - mean;
  float q = d0 * d0 + d1 * d1 + d2 * d2;
#pragma unroll
  for (int m = 16; m; m >>= 1) q += __shfl_xor(q, m, 32);
  float inv = rsqrtf(q * (1.0f / 96.0f) + 1e-5f);
  size_t row;
  if (doShift) {
    int bi = t / 3136, hw = t % 3136, h = hw / 56, w = hw % 56;
    int h2 = (h + 53) % 56, w2 = (w + 53) % 56;  // roll by -3
    row = ((size_t)bi * 64 + (h2 / 7) * 8 + (w2 / 7)) * 49 + (h2 % 7) * 7 + (w2 % 7);
  } else {
    row = (size_t)t;
  }
  half_t* op = out + row * 96;
  op[lane]      = (half_t)(d0 * inv * g[lane]      + b[lane]);
  op[lane + 32] = (half_t)(d1 * inv * g[lane + 32] + b[lane + 32]);
  op[lane + 64] = (half_t)(d2 * inv * g[lane + 64] + b[lane + 64]);
}

// ---------------- unified f16 WMMA GEMM ------------------------------------------------
// D(M x Ncols) = epilogue(A(M x KDIM) @ W(KDIM x Ncols) + bias)
// block = 8 waves; weight tile (16 cols x KDIM) staged TRANSPOSED in LDS once per block;
// each wave computes TWO 16x16 D tiles (32 rows), reusing one B fragment per K-step.
// grid = (Ncols/16, M/256)
// MODE 0: f16 out           (qkv)
// MODE 1: f16 out + GELU    (fc1)
// MODE 2: f32 out, window-reverse + un-shift + residual xres -> xout   (proj)
// MODE 3: f32 accumulate into xout (holds x1)                          (fc2)
template <int KDIM, int MODE>
__global__ __launch_bounds__(256) void gemm_f16_kernel(const half_t* __restrict__ A,
                                                       const half_t* __restrict__ W,
                                                       const float* __restrict__ bias,
                                                       half_t* __restrict__ Out,
                                                       const float* __restrict__ xres,
                                                       float* __restrict__ xout,
                                                       int Ncols) {
  __shared__ __align__(16) half_t Wt[16 * KDIM];  // transposed: Wt[n][k]
  int lane = threadIdx.x & 31, wid = threadIdx.x >> 5;
  int hi = lane >> 4, l16 = lane & 15;
  int nbase = blockIdx.x * 16;
  int ncol = nbase + l16;

  // ---- cooperative stage + transpose of the weight tile (coalesced global reads) ----
#pragma unroll 1
  for (int i = threadIdx.x; i < 16 * KDIM; i += 256) {
    int k = i >> 4, n = i & 15;
    Wt[n * KDIM + k] = W[(size_t)k * Ncols + nbase + n];
  }
  __syncthreads();

  int mbase = (blockIdx.y * 8 + wid) * 32;  // two 16-row tiles per wave
  int arow0 = mbase + l16, arow1 = mbase + 16 + l16;
  __builtin_prefetch(A + (size_t)arow0 * KDIM, 0, 3);  // global_prefetch_b8
  __builtin_prefetch(A + (size_t)arow1 * KDIM, 0, 3);

  v8f acc0 = {}, acc1 = {};
#pragma unroll
  for (int kk = 0; kk < KDIM; kk += 32) {
    const v8h* pa0 = (const v8h*)(A + (size_t)arow0 * KDIM + kk + hi * 8);
    const v8h* pa1 = (const v8h*)(A + (size_t)arow1 * KDIM + kk + hi * 8);
    v16h af0 = combine8(pa0[0], pa0[2]);
    v16h af1 = combine8(pa1[0], pa1[2]);
    const v8h* pb = (const v8h*)(Wt + l16 * KDIM + kk + hi * 8);  // table-order K
    v16h bf = combine8(pb[0], pb[2]);
    acc0 = wmma16(af0, bf, acc0);
    acc1 = wmma16(af1, bf, acc1);
  }

  float bv = bias[ncol];
#pragma unroll
  for (int sub = 0; sub < 2; ++sub) {
    v8f acc = sub ? acc1 : acc0;
#pragma unroll
    for (int g = 0; g < 8; ++g) {
      int r = mbase + sub * 16 + g + hi * 8;
      float v = acc[g] + bv;
      if (MODE == 0) {
        Out[(size_t)r * Ncols + ncol] = (half_t)v;
      } else if (MODE == 1) {
        v = 0.5f * v * (1.0f + erff(v * 0.70710678118f));  // exact GELU
        Out[(size_t)r * Ncols + ncol] = (half_t)v;
      } else if (MODE == 2) {
        int window = r / 49, tok = r % 49;  // r is window-major token
        int bi = window / 64, wl = window % 64;
        int hh = (wl / 8) * 7 + tok / 7, ww = (wl % 8) * 7 + tok % 7;
        int h = (hh + 3) % 56, w = (ww + 3) % 56;  // un-roll (+3)
        size_t idx = ((size_t)bi * 3136 + h * 56 + w) * 96 + ncol;
        xout[idx] = xres[idx] + v;  // x1 = shortcut + proj(attn)
      } else {  // MODE 3
        size_t idx = (size_t)r * 96 + ncol;
        xout[idx] = xout[idx] + v;  // y = x1 + mlp
      }
    }
  }
}

// ---------------- windowed attention: 1 block = (window, head), 4 waves -------------
// QKV layout: row = window*49+token, col = qkv*96 + head*32 + d   (f16)
__global__ __launch_bounds__(128) void attn_kernel(const half_t* __restrict__ QKV,
                                                   const float* __restrict__ rpb,
                                                   half_t* __restrict__ O) {
  __shared__ __align__(32) half_t P[64 * 64];
  int lane = threadIdx.x & 31, wid = threadIdx.x >> 5;
  int hi = lane >> 4, l16 = lane & 15;
  int window = blockIdx.x / 3, head = blockIdx.x % 3;
  int wl = window & 63, winh = wl >> 3, winw = wl & 7;
  size_t base = (size_t)window * 49 * 288;
  int mbase = wid * 16;

  // ---- S = (q @ k^T) : wave does rows [mbase, mbase+16) x all 64 cols (4 tiles) ----
  v16h qf = {};
  int qrow = mbase + l16;
  if (qrow < 49) {
    const v8h* pq = (const v8h*)(QKV + base + (size_t)qrow * 288 + head * 32 + hi * 8);
    qf = combine8(pq[0], pq[2]);
  }
  v8f s[4];
#pragma unroll
  for (int nt = 0; nt < 4; ++nt) {
    v16h kf = {};
    int j = nt * 16 + l16;
    if (j < 49) {  // k^T column j = contiguous 32 halves of key j
      const v8h* pk = (const v8h*)(QKV + base + (size_t)j * 288 + 96 + head * 32 + hi * 8);
      kf = combine8(pk[0], pk[2]);
    }
    v8f z = {};
    s[nt] = wmma16(qf, kf, z);
  }

  // ---- scale + relative-position bias + shift mask (computed analytically) ----
#pragma unroll
  for (int nt = 0; nt < 4; ++nt) {
    int n = nt * 16 + l16;
    int nh2 = n / 7, nw2 = n % 7;
    int idn = regionOf(winh * 7 + nh2) * 3 + regionOf(winw * 7 + nw2);
#pragma unroll
    for (int g = 0; g < 8; ++g) {
      int m = mbase + g + hi * 8;
      if (m < 49 && n < 49) {
        int mh = m / 7, mw = m % 7;
        int rel = (mh - nh2 + 6) * 13 + (mw - nw2 + 6);
        float bias = rpb[rel * 3 + head];
        int idm = regionOf(winh * 7 + mh) * 3 + regionOf(winw * 7 + mw);
        float msk = (idm != idn) ? -100.0f : 0.0f;
        s[nt][g] = s[nt][g] * SCALE + bias + msk;
      } else {
        s[nt][g] = -1e30f;  // padding -> softmax weight 0
      }
    }
  }

  // ---- row softmax: row lives across lanes (l&15) of one half-wave x 4 tiles ----
#pragma unroll
  for (int g = 0; g < 8; ++g) {
    float mx = s[0][g];
#pragma unroll
    for (int nt = 1; nt < 4; ++nt) mx = fmaxf(mx, s[nt][g]);
#pragma unroll
    for (int m = 8; m; m >>= 1) mx = fmaxf(mx, __shfl_xor(mx, m, 32));
    float sum = 0.0f;
#pragma unroll
    for (int nt = 0; nt < 4; ++nt) {
      float e = __expf(s[nt][g] - mx);
      s[nt][g] = e;
      sum += e;
    }
#pragma unroll
    for (int m = 8; m; m >>= 1) sum += __shfl_xor(sum, m, 32);
    float inv = 1.0f / sum;
#pragma unroll
    for (int nt = 0; nt < 4; ++nt) s[nt][g] *= inv;
  }

  // ---- re-stripe P through LDS into A-fragment layout ----
#pragma unroll
  for (int nt = 0; nt < 4; ++nt)
#pragma unroll
    for (int g = 0; g < 8; ++g)
      P[(mbase + g + hi * 8) * 64 + nt * 16 + l16] = (half_t)s[nt][g];
  __syncthreads();

  // ---- O = P @ V : wave -> rows [mbase, mbase+16) x 32 dims (2 tiles) ----
  v8f o0 = {}, o1 = {};
#pragma unroll
  for (int kk = 0; kk < 64; kk += 32) {
    const v8h* pp = (const v8h*)(P + (mbase + l16) * 64 + kk + hi * 8);
    v16h pf = combine8(pp[0], pp[2]);
#pragma unroll
    for (int dt = 0; dt < 2; ++dt) {
      v16h vf = {};
      int d = dt * 16 + l16;
#pragma unroll
      for (int h2 = 0; h2 < 16; ++h2) {
        int j = kk + (h2 & 7) + hi * 8 + (h2 >> 3) * 16;
        vf[h2] = (j < 49) ? QKV[base + (size_t)j * 288 + 192 + head * 32 + d] : (half_t)0;
      }
      if (dt == 0) o0 = wmma16(pf, vf, o0);
      else         o1 = wmma16(pf, vf, o1);
    }
  }
#pragma unroll
  for (int g = 0; g < 8; ++g) {
    int m = mbase + g + hi * 8;
    if (m < 49) {
      size_t orow = ((size_t)window * 49 + m) * 96 + head * 32;
      O[orow + l16]      = (half_t)o0[g];
      O[orow + 16 + l16] = (half_t)o1[g];
    }
  }
}

// ---------------- launcher ----------------
extern "C" void kernel_launch(void* const* d_in, const int* in_sizes, int n_in,
                              void* d_out, int out_size, void* d_ws, size_t ws_size,
                              hipStream_t stream) {
  const float* x      = (const float*)d_in[0];
  const float* n1g    = (const float*)d_in[1];
  const float* n1b    = (const float*)d_in[2];
  const float* qkv_w  = (const float*)d_in[3];
  const float* qkv_b  = (const float*)d_in[4];
  const float* rpb    = (const float*)d_in[5];
  const float* proj_w = (const float*)d_in[6];
  const float* proj_b = (const float*)d_in[7];
  const float* n2g    = (const float*)d_in[8];
  const float* n2b    = (const float*)d_in[9];
  const float* fc1_w  = (const float*)d_in[10];
  const float* fc1_b  = (const float*)d_in[11];
  const float* fc2_w  = (const float*)d_in[12];
  const float* fc2_b  = (const float*)d_in[13];
  float* out = (float*)d_out;

  char* ws = (char*)d_ws;
  half_t* w_qkv16  = (half_t*)(ws + 0);          // 27648 halves
  half_t* w_proj16 = (half_t*)(ws + 55296);      // 9216 halves
  half_t* w_fc116  = (half_t*)(ws + 73728);      // 36864 halves
  half_t* w_fc216  = (half_t*)(ws + 147456);     // 36864 halves
  half_t* bufA     = (half_t*)(ws + 262144);                // TOK x 96  f16 (77.1 MB)
  half_t* bufQKV   = (half_t*)(ws + 77332480ull);           // TOK x 288 f16 (231 MB)
  half_t* bufO     = (half_t*)(ws + 308543488ull);          // TOK x 96  f16 (77.1 MB)
  half_t* bufH     = (half_t*)(ws + 77332480ull);           // TOK x 384 f16, reuses QKV+O

  // 1) weights -> f16
  cvt_f16_kernel<<<(27648 + 255) / 256, 256, 0, stream>>>(qkv_w, w_qkv16, 27648);
  cvt_f16_kernel<<<(9216  + 255) / 256, 256, 0, stream>>>(proj_w, w_proj16, 9216);
  cvt_f16_kernel<<<(36864 + 255) / 256, 256, 0, stream>>>(fc1_w, w_fc116, 36864);
  cvt_f16_kernel<<<(36864 + 255) / 256, 256, 0, stream>>>(fc2_w, w_fc216, 36864);

  // 2) LN1 + roll + window partition -> f16 (window-major rows)
  ln_kernel<<<TOK / 4, 128, 0, stream>>>(x, n1g, n1b, bufA, 1);

  // 3) qkv GEMM: (401408 x 96) @ (96 x 288)   [MODE 0]
  gemm_f16_kernel<96, 0><<<dim3(18, 1568), 256, 0, stream>>>(
      bufA, w_qkv16, qkv_b, bufQKV, nullptr, nullptr, 288);

  // 4) windowed attention, per (window, head)
  attn_kernel<<<NWIN * 3, 128, 0, stream>>>(bufQKV, rpb, bufO);

  // 5) proj GEMM + window reverse + un-shift + residual -> d_out (x1, f32)  [MODE 2]
  gemm_f16_kernel<96, 2><<<dim3(6, 1568), 256, 0, stream>>>(
      bufO, w_proj16, proj_b, nullptr, x, out, 96);

  // 6) LN2 (natural token order) -> f16
  ln_kernel<<<TOK / 4, 128, 0, stream>>>(out, n2g, n2b, bufA, 0);

  // 7) fc1 GEMM + exact GELU: (401408 x 96) @ (96 x 384)  [MODE 1]
  gemm_f16_kernel<96, 1><<<dim3(24, 1568), 256, 0, stream>>>(
      bufA, w_fc116, fc1_b, bufH, nullptr, nullptr, 384);

  // 8) fc2 GEMM + residual: (401408 x 384) @ (384 x 96) onto x1 in d_out  [MODE 3]
  gemm_f16_kernel<384, 3><<<dim3(6, 1568), 256, 0, stream>>>(
      bufH, w_fc216, fc2_b, nullptr, nullptr, out, 96);
}
